// LSTMModel_23656679867573
// MI455X (gfx1250) — matrix operand order repaired and checked by reading, not compiled, not run on hardware
//
#include <hip/hip_runtime.h>

typedef __attribute__((ext_vector_type(16))) __bf16 v16bf;
typedef __attribute__((ext_vector_type(8)))  __bf16 v8bf;
typedef __attribute__((ext_vector_type(8)))  float  v8f;

#define XCATP 288   // 263 padded to multiple of 32

__device__ __forceinline__ __bf16 f2bf(float x) { return (__bf16)x; }

// load 8 consecutive f32, convert to bf16 into fragment slots [o..o+7]
__device__ __forceinline__ void load8f_bf(const float* __restrict__ p, v16bf& d, int o) {
  float4 f0 = *(const float4*)(p);
  float4 f1 = *(const float4*)(p + 4);
  d[o+0]=f2bf(f0.x); d[o+1]=f2bf(f0.y); d[o+2]=f2bf(f0.z); d[o+3]=f2bf(f0.w);
  d[o+4]=f2bf(f1.x); d[o+5]=f2bf(f1.y); d[o+6]=f2bf(f1.z); d[o+7]=f2bf(f1.w);
}

// load 8 consecutive bf16 into fragment slots [o..o+7]
__device__ __forceinline__ void load8bf(const __bf16* __restrict__ p, v16bf& d, int o) {
  v8bf t = *(const v8bf*)(p);
#pragma unroll
  for (int i = 0; i < 8; ++i) d[o+i] = t[i];
}

__device__ __forceinline__ v8f wmma_bf16(v16bf a, v16bf b, v8f c) {
  return __builtin_amdgcn_wmma_f32_16x16x32_bf16(false, a, false, b, (short)0, c, false, false);
}

__device__ __forceinline__ float sigmoidf(float x) { return 1.0f / (1.0f + __expf(-x)); }

// ---------------------------------------------------------------------------
// conv1: obs [2048,4,84,84] f32, w [16,256] -> out NCHW bf16 [n][16][400]
// one wave = 16 pixels x 16 out-channels, K = 4*8*8 = 256
// (HBM-bound: 231 MB obs read; no tile sharing possible with only 16 channels)
// ---------------------------------------------------------------------------
__global__ void __launch_bounds__(32) conv1_kernel(const float* __restrict__ obs,
    const float* __restrict__ w, const float* __restrict__ bias,
    __bf16* __restrict__ out) {
  int wid = blockIdx.x;               // n*25 + ptile
  int n = wid / 25, ptile = wid % 25;
  int lane = threadIdx.x;
  int hh = lane >> 4;
  int p = ptile * 16 + (lane & 15);   // this lane's A row (output pixel)
  int oy = p / 20, ox = p % 20;
  int iy = oy * 4, ix = ox * 4;
  const float* obsn = obs + (size_t)n * (4 * 84 * 84);
  const float* wrow = w + (lane & 15) * 256;   // B column = out channel
  v8f acc = {};
#pragma unroll
  for (int kk = 0; kk < 8; ++kk) {
    v16bf a, b;
#pragma unroll
    for (int g = 0; g < 2; ++g) {
      int kbase = kk * 32 + 8 * hh + 16 * g;   // 8 consecutive k = one 8-wide row
      int c = kbase >> 6;
      int r = (kbase >> 3) & 7;
      load8f_bf(obsn + (c * 84 + iy + r) * 84 + ix, a, g * 8);
    }
    int kbase = kk * 32 + 8 * hh;
    load8f_bf(wrow + kbase,      b, 0);
    load8f_bf(wrow + kbase + 16, b, 8);
    acc = wmma_bf16(a, b, acc);
  }
  int ch = lane & 15;
  float bv = bias[ch];
  __bf16* dst = out + ((size_t)n * 16 + ch) * 400 + ptile * 16 + 8 * hh;
#pragma unroll
  for (int v = 0; v < 8; v += 2) {
    float x0 = acc[v]   + bv; x0 = x0 > 0.f ? x0 : 0.f;
    float x1 = acc[v+1] + bv; x1 = x1 > 0.f ? x1 : 0.f;
    union { __bf16 h[2]; unsigned u; } pk;
    pk.h[0] = f2bf(x0); pk.h[1] = f2bf(x1);
    *(unsigned*)(dst + v) = pk.u;              // rows contiguous -> packed b32 store
  }
}

// ---------------------------------------------------------------------------
// conv2: in NCHW bf16 [n][16][20][20], w [32,256] -> out bf16 [n][2592]
// one wave = 16 pixels x ALL 32 out-channels (two N-tiles share the scattered
// A gather; the two WMMAs are independent -> hides XDL latency, no hazard nops)
// ---------------------------------------------------------------------------
__global__ void __launch_bounds__(32) conv2_kernel(const __bf16* __restrict__ in,
    const float* __restrict__ w, const float* __restrict__ bias,
    __bf16* __restrict__ out) {
  int wid = blockIdx.x;               // n*6 + ptile
  int n = wid / 6; int ptile = wid % 6;
  int lane = threadIdx.x; int hh = lane >> 4;
  int p = ptile * 16 + (lane & 15);
  bool pv = p < 81;
  int pc = pv ? p : 0;
  int oy = pc / 9, ox = pc % 9;
  int iy = oy * 2, ix = ox * 2;
  const __bf16* inn = in + (size_t)n * 16 * 400;
  int nc = lane & 15;
  const float* wrow0 = w + nc * 256;            // channels 0..15
  const float* wrow1 = w + (16 + nc) * 256;     // channels 16..31
  v8f acc0 = {}, acc1 = {};
#pragma unroll
  for (int kk = 0; kk < 8; ++kk) {
    v16bf a, b0, b1;
#pragma unroll
    for (int g = 0; g < 2; ++g) {
      int kbase = kk * 32 + 8 * hh + 16 * g;   // k = c*16 + r*4 + s
      int c  = kbase >> 4;
      int r0 = (kbase >> 2) & 3;               // rows r0, r0+1, s = 0..3 each
      const __bf16* s0 = inn + (c * 20 + iy + r0    ) * 20 + ix;
      const __bf16* s1 = inn + (c * 20 + iy + r0 + 1) * 20 + ix;
#pragma unroll
      for (int s = 0; s < 4; ++s) a[g*8 + s]     = pv ? s0[s] : f2bf(0.f);
#pragma unroll
      for (int s = 0; s < 4; ++s) a[g*8 + 4 + s] = pv ? s1[s] : f2bf(0.f);
    }
    int kbase = kk * 32 + 8 * hh;
    load8f_bf(wrow0 + kbase,      b0, 0);
    load8f_bf(wrow0 + kbase + 16, b0, 8);
    load8f_bf(wrow1 + kbase,      b1, 0);
    load8f_bf(wrow1 + kbase + 16, b1, 8);
    acc0 = wmma_bf16(a, b0, acc0);
    acc1 = wmma_bf16(a, b1, acc1);
  }
  float bv0 = bias[nc];
  float bv1 = bias[16 + nc];
  __bf16* dst0 = out + (size_t)n * 2592 + nc * 81;
  __bf16* dst1 = out + (size_t)n * 2592 + (16 + nc) * 81;
#pragma unroll
  for (int v = 0; v < 8; ++v) {
    int p2 = ptile * 16 + v + 8 * hh;
    if (p2 < 81) {
      float x0 = acc0[v] + bv0; x0 = x0 > 0.f ? x0 : 0.f;
      float x1 = acc1[v] + bv1; x1 = x1 > 0.f ? x1 : 0.f;
      dst0[p2] = f2bf(x0);
      dst1[p2] = f2bf(x1);
    }
  }
}

// ---------------------------------------------------------------------------
// fc: A bf16 [2048][2592] x fc_w[256][2592]^T -> relu -> xcat[:, 0:256]
// one wave = TWO 16-row M-tiles sharing the B (weight) fragment: halves the
// f32->bf16 weight conversion per WMMA and interleaves independent WMMAs.
// ---------------------------------------------------------------------------
__global__ void __launch_bounds__(32) fc_kernel(const __bf16* __restrict__ A,
    const float* __restrict__ W, const float* __restrict__ bias,
    __bf16* __restrict__ xcat) {
  int wid = blockIdx.x;               // mpair*16 + ntile  (64 mpairs x 16 ntiles)
  int mpair = wid >> 4, ntile = wid & 15;
  int lane = threadIdx.x; int hh = lane >> 4;
  int ncol = ntile * 16 + (lane & 15);
  const __bf16* arow0 = A + (size_t)(mpair * 32 + (lane & 15)) * 2592;
  const __bf16* arow1 = arow0 + (size_t)16 * 2592;
  const float*  wrow  = W + (size_t)ncol * 2592;
  v8f acc0 = {}, acc1 = {};
  for (int kk = 0; kk < 81; ++kk) {
    v16bf a0, a1, b;
    int kb0 = kk * 32 + 8 * hh;
    load8bf(arow0 + kb0,      a0, 0);
    load8bf(arow0 + kb0 + 16, a0, 8);
    load8bf(arow1 + kb0,      a1, 0);
    load8bf(arow1 + kb0 + 16, a1, 8);
    int kbase = kk * 32 + 8 * hh;
    __builtin_prefetch(wrow + kbase + 32, 0, 1);   // global_prefetch_b8
    load8f_bf(wrow + kbase,      b, 0);
    load8f_bf(wrow + kbase + 16, b, 8);
    acc0 = wmma_bf16(a0, b, acc0);
    acc1 = wmma_bf16(a1, b, acc1);
  }
  float bv = bias[ncol];
#pragma unroll
  for (int v = 0; v < 8; ++v) {
    int row0 = mpair * 32 + v + 8 * hh;
    float x0 = acc0[v] + bv; x0 = x0 > 0.f ? x0 : 0.f;
    float x1 = acc1[v] + bv; x1 = x1 > 0.f ? x1 : 0.f;
    xcat[(size_t)row0 * XCATP + ncol] = f2bf(x0);
    xcat[(size_t)(row0 + 16) * XCATP + ncol] = f2bf(x1);
  }
}

// xcat columns 256..287: last_reward_action (7) + zero pad
__global__ void xcat_fill_kernel(const float* __restrict__ lra, __bf16* __restrict__ xcat) {
  int row = blockIdx.x; int c = threadIdx.x;      // 32 threads
  float v = lra[row * 7 + (c < 7 ? c : 0)] * (c < 7 ? 1.f : 0.f);
  xcat[(size_t)row * XCATP + 256 + c] = f2bf(v);
}

// ---------------------------------------------------------------------------
// gates_x = xcat[2048][288(263)] @ w_ih[1024][263]^T + b_ih + b_hh -> f32 [2048][1024]
// one wave = TWO M-tiles sharing B (w_ih has stride 263 -> scalar clamped loads,
// so amortizing them across two tiles matters most here).
// ---------------------------------------------------------------------------
__global__ void __launch_bounds__(32) gatesx_kernel(const __bf16* __restrict__ xcat,
    const float* __restrict__ wih, const float* __restrict__ bih,
    const float* __restrict__ bhh, float* __restrict__ gx) {
  int wid = blockIdx.x;               // mpair*64 + ntile  (64 mpairs x 64 ntiles)
  int mpair = wid >> 6, ntile = wid & 63;
  int lane = threadIdx.x; int hh = lane >> 4;
  int ncol = ntile * 16 + (lane & 15);
  const __bf16* arow0 = xcat + (size_t)(mpair * 32 + (lane & 15)) * XCATP;
  const __bf16* arow1 = arow0 + (size_t)16 * XCATP;
  const float*  wrow  = wih + (size_t)ncol * 263;
  v8f acc0 = {}, acc1 = {};
#pragma unroll
  for (int kk = 0; kk < 9; ++kk) {
    v16bf a0, a1, b;
    int kb0 = kk * 32 + 8 * hh;
    load8bf(arow0 + kb0,      a0, 0);
    load8bf(arow0 + kb0 + 16, a0, 8);
    load8bf(arow1 + kb0,      a1, 0);
    load8bf(arow1 + kb0 + 16, a1, 8);
    int kbase = kk * 32 + 8 * hh;
#pragma unroll
    for (int j = 0; j < 16; ++j) {
      int k = kbase + (j & 7) + 16 * (j >> 3);
      float wv = wrow[k < 263 ? k : 0] * (k < 263 ? 1.f : 0.f);  // clamp: no OOB speculation
      b[j] = f2bf(wv);
    }
    acc0 = wmma_bf16(a0, b, acc0);
    acc1 = wmma_bf16(a1, b, acc1);
  }
  float bsum = bih[ncol] + bhh[ncol];
#pragma unroll
  for (int v = 0; v < 8; ++v) {
    int row0 = mpair * 32 + v + 8 * hh;
    gx[(size_t)row0 * 1024 + ncol] = acc0[v] + bsum;
    gx[(size_t)(row0 + 16) * 1024 + ncol] = acc1[v] + bsum;
  }
}

// ---------------------------------------------------------------------------
// Sequential LSTM: persistent 16 workgroups (cluster-sized), each owns 16 hidden
// units (its 4x16 gate columns). W_hh slice lives in LDS as bf16. Per step:
// grid barrier -> stage masked h (LDS) -> grid barrier -> WMMA h@Whh^T -> cell.
// ---------------------------------------------------------------------------
__device__ __forceinline__ void grid_barrier(unsigned* bar, int phase) {
  __syncthreads();
  __threadfence();
  if (threadIdx.x == 0) {
    unsigned target = (unsigned)phase * 16u;
    unsigned arrived = __hip_atomic_fetch_add(&bar[0], 1u, __ATOMIC_ACQ_REL,
                                              __HIP_MEMORY_SCOPE_AGENT) + 1u;
    if (arrived == target) {
      __hip_atomic_store(&bar[1], target, __ATOMIC_RELEASE, __HIP_MEMORY_SCOPE_AGENT);
    } else {
      while (__hip_atomic_load(&bar[1], __ATOMIC_ACQUIRE, __HIP_MEMORY_SCOPE_AGENT) < target)
        __builtin_amdgcn_s_sleep(1);
    }
  }
  __syncthreads();
  __builtin_amdgcn_s_cluster_barrier();  // s_barrier_signal/wait -3 (NOP if not a cluster)
}

__global__ void __launch_bounds__(128) lstm_kernel(
    const float* __restrict__ gx, const float* __restrict__ masks,
    const float* __restrict__ h0, const float* __restrict__ c0,
    const float* __restrict__ whh_g, float* __restrict__ hs,
    float* __restrict__ hbuf, unsigned* __restrict__ bar,
    float* __restrict__ outH, float* __restrict__ outC) {
  __shared__ __bf16 whh[64][256];   // this WG's W_hh slice (4 gates x 16 units, K=256)
  __shared__ __bf16 hm[16][256];    // masked h (bf16) for A fragments
  __shared__ float  gbuf[64][16];   // GEMM result [nLocal][batch]
  __shared__ float  cst[16][16];    // c state [batch][unit]
  __shared__ float  mt[16];
  int blk = blockIdx.x;             // 0..15
  int tid = threadIdx.x;            // 0..127 (4 waves)
  int lane = tid & 31; int wv = tid >> 5;
  int hh = lane >> 4;

  // stage W_hh slice: nLocal = gate*16 + unit -> global row gate*256 + blk*16 + unit
  for (int i = tid; i < 64 * 256; i += 128) {
    int nl = i >> 8, k = i & 255;
    int gate = nl >> 4, u = nl & 15;
    int ng = gate * 256 + blk * 16 + u;
    whh[nl][k] = f2bf(whh_g[(size_t)ng * 256 + k]);
  }
  // init c state and publish h0 slice
  for (int i = tid; i < 256; i += 128) {
    int b = i >> 4, u = i & 15;
    cst[b][u] = c0[b * 256 + blk * 16 + u];
    __hip_atomic_store(&hbuf[b * 256 + blk * 16 + u], h0[b * 256 + blk * 16 + u],
                       __ATOMIC_RELAXED, __HIP_MEMORY_SCOPE_AGENT);
  }

  for (int t = 0; t < 128; ++t) {
    grid_barrier(bar, 2 * t + 1);           // prev-step h writes visible everywhere
    if (tid < 16) mt[tid] = masks[tid * 128 + t];
    __syncthreads();
    for (int i = tid; i < 16 * 256; i += 128) {
      int b = i >> 8, k = i & 255;
      float hv = __hip_atomic_load(&hbuf[b * 256 + k], __ATOMIC_RELAXED,
                                   __HIP_MEMORY_SCOPE_AGENT);
      hm[b][k] = f2bf(hv * mt[b]);
    }
    grid_barrier(bar, 2 * t + 2);           // all blocks done reading hbuf

    // gates_h = hm(16x256) @ whh_slice^T -> 16x64; one 16x16 ntile per wave
    {
      v8f acc = {};
      int nl = wv * 16 + (lane & 15);
      const __bf16* brow = &whh[nl][0];
      const __bf16* arow = &hm[lane & 15][0];
#pragma unroll
      for (int kk = 0; kk < 8; ++kk) {
        v16bf a, b;
        int kb0 = kk * 32 + 8 * hh;
        load8bf(arow + kb0,      a, 0);
        load8bf(arow + kb0 + 16, a, 8);
        int kb1 = kk * 32 + 8 * hh;
        load8bf(brow + kb1,      b, 0);
        load8bf(brow + kb1 + 16, b, 8);
        acc = wmma_bf16(a, b, acc);
      }
#pragma unroll
      for (int v = 0; v < 8; ++v) gbuf[nl][v + 8 * hh] = acc[v];
    }
    __syncthreads();

    // LSTM cell for this block's 16 units x 16 batches
    for (int i = tid; i < 256; i += 128) {
      int b = i >> 4, u = i & 15;
      const float* gxr = gx + (size_t)(b * 128 + t) * 1024 + blk * 16 + u;
      float iv = gbuf[ 0 + u][b] + gxr[0];
      float fv = gbuf[16 + u][b] + gxr[256];
      float gv = gbuf[32 + u][b] + gxr[512];
      float ov = gbuf[48 + u][b] + gxr[768];
      float cold = cst[b][u] * mt[b];
      float cn = sigmoidf(fv) * cold + sigmoidf(iv) * tanhf(gv);
      float hn = sigmoidf(ov) * tanhf(cn);
      cst[b][u] = cn;
      hs[(size_t)(b * 128 + t) * 256 + blk * 16 + u] = hn;
      __hip_atomic_store(&hbuf[b * 256 + blk * 16 + u], hn,
                         __ATOMIC_RELAXED, __HIP_MEMORY_SCOPE_AGENT);
      if (t == 127) {
        outH[b * 256 + blk * 16 + u] = hn;
        outC[b * 256 + blk * 16 + u] = cn;
      }
    }
  }
}

// policy (6) + critic (1) heads: one thread per (row, head-col)
__global__ void heads_kernel(const float* __restrict__ hs,
    const float* __restrict__ pw, const float* __restrict__ pb,
    const float* __restrict__ cw, const float* __restrict__ cb,
    float* __restrict__ outP, float* __restrict__ outC) {
  int idx = blockIdx.x * blockDim.x + threadIdx.x;
  if (idx >= 2048 * 7) return;
  int row = idx / 7, a = idx % 7;
  const float* f = hs + (size_t)row * 256;
  const float* w = (a < 6) ? (pw + a * 256) : cw;
  float acc = 0.f;
#pragma unroll 8
  for (int k = 0; k < 256; ++k) acc += f[k] * w[k];
  if (a < 6) outP[row * 6 + a] = acc + pb[a];
  else       outC[row] = acc + cb[0];
}

__global__ void init_kernel(unsigned* bar) {
  if (threadIdx.x == 0) { bar[0] = 0u; bar[1] = 0u; }
}

extern "C" void kernel_launch(void* const* d_in, const int* in_sizes, int n_in,
                              void* d_out, int out_size, void* d_ws, size_t ws_size,
                              hipStream_t stream) {
  (void)in_sizes; (void)n_in; (void)out_size; (void)ws_size;
  const float* obs   = (const float*)d_in[0];
  const float* lra   = (const float*)d_in[1];
  const float* masks = (const float*)d_in[2];
  const float* h0    = (const float*)d_in[3];
  const float* c0    = (const float*)d_in[4];
  const float* c1w   = (const float*)d_in[5];
  const float* c1b   = (const float*)d_in[6];
  const float* c2w   = (const float*)d_in[7];
  const float* c2b   = (const float*)d_in[8];
  const float* fcw   = (const float*)d_in[9];
  const float* fcb   = (const float*)d_in[10];
  const float* wih   = (const float*)d_in[11];
  const float* whh   = (const float*)d_in[12];
  const float* bih   = (const float*)d_in[13];
  const float* bhh   = (const float*)d_in[14];
  const float* pw    = (const float*)d_in[15];
  const float* pb    = (const float*)d_in[16];
  const float* cwt   = (const float*)d_in[17];
  const float* cb    = (const float*)d_in[18];

  char* ws = (char*)d_ws;
  size_t off = 0;
  auto wsa = [&](size_t bytes) {
    void* p = (void*)(ws + off);
    off = (off + bytes + 255) & ~(size_t)255;
    return p;
  };
  __bf16*   conv1out = (__bf16*)  wsa((size_t)2048 * 16 * 400 * 2); // 26.2 MB, NCHW bf16
  __bf16*   conv2out = (__bf16*)  wsa((size_t)2048 * 2592 * 2);     // 10.6 MB, flatten order
  __bf16*   xcat     = (__bf16*)  wsa((size_t)2048 * XCATP * 2);    // 1.2 MB
  float*    gx       = (float*)   wsa((size_t)2048 * 1024 * 4);     // 8.4 MB
  float*    hs       = (float*)   wsa((size_t)2048 * 256 * 4);      // 2.1 MB
  float*    hbuf     = (float*)   wsa((size_t)16 * 256 * 4);        // h exchange
  unsigned* bar      = (unsigned*)wsa(256);                          // grid barrier

  float* out   = (float*)d_out;
  float* outP  = out;             // [16,128,6]  = 12288
  float* outCr = out + 12288;     // [16,128,1]  =  2048
  float* outH  = out + 14336;     // [16,1,256]  =  4096
  float* outC  = out + 18432;     // [16,1,256]  =  4096

  init_kernel<<<1, 32, 0, stream>>>(bar);
  conv1_kernel<<<2048 * 25, 32, 0, stream>>>(obs, c1w, c1b, conv1out);
  conv2_kernel<<<2048 * 6, 32, 0, stream>>>(conv1out, c2w, c2b, conv2out);
  fc_kernel<<<64 * 16, 32, 0, stream>>>(conv2out, fcw, fcb, xcat);
  xcat_fill_kernel<<<2048, 32, 0, stream>>>(lra, xcat);
  gatesx_kernel<<<64 * 64, 32, 0, stream>>>(xcat, wih, bih, bhh, gx);
  lstm_kernel<<<16, 128, 0, stream>>>(gx, masks, h0, c0, whh, hs, hbuf, bar, outH, outC);
  heads_kernel<<<(2048 * 7 + 255) / 256, 256, 0, stream>>>(hs, pw, pb, cwt, cb, outP, outCr);
}
